// MultiHeadAttention_20744692040179
// MI455X (gfx1250) — compile-verified, matrix operand order checked by reading, not run on hardware
//
#include <hip/hip_runtime.h>
#include <hip/hip_bf16.h>

// Sizes from the reference
#define BB 2
#define SS 2048
#define DD 256
#define HH 8
#define DKV 32
#define LN_EPS 1e-5f

typedef _Float16 h16;
typedef __attribute__((ext_vector_type(16))) _Float16 v16h;
typedef __attribute__((ext_vector_type(8)))  _Float16 v8h;
typedef __attribute__((ext_vector_type(8)))  float    v8f;

// 16-bit A/B fragment element map (ISA 7.12.2), lane: idx = lane&15, half = lane>>4
//   elements 0..7  -> K = half*8 + e        (contiguous run of 8)
//   elements 8..15 -> K = 16 + half*8 + e-8 (contiguous run of 8)
// => fragment = two 16-byte contiguous chunks when the 32 K-values are contiguous.
__device__ __forceinline__ v16h frag_h16(const h16* rowptr, int half) {
    v8h lo = *(const v8h*)(rowptr + half * 8);
    v8h hi = *(const v8h*)(rowptr + 16 + half * 8);
    return __builtin_shufflevector(lo, hi, 0, 1, 2, 3, 4, 5, 6, 7,
                                   8, 9, 10, 11, 12, 13, 14, 15);
}

__device__ __forceinline__ v16h frag_f32(const float* rowptr, int half) {
    float4 a0 = *(const float4*)(rowptr + half * 8);
    float4 a1 = *(const float4*)(rowptr + half * 8 + 4);
    float4 a2 = *(const float4*)(rowptr + 16 + half * 8);
    float4 a3 = *(const float4*)(rowptr + 16 + half * 8 + 4);
    v16h r;
    r[0] = (h16)a0.x; r[1] = (h16)a0.y; r[2]  = (h16)a0.z; r[3]  = (h16)a0.w;
    r[4] = (h16)a1.x; r[5] = (h16)a1.y; r[6]  = (h16)a1.z; r[7]  = (h16)a1.w;
    r[8] = (h16)a2.x; r[9] = (h16)a2.y; r[10] = (h16)a2.z; r[11] = (h16)a2.w;
    r[12] = (h16)a3.x; r[13] = (h16)a3.y; r[14] = (h16)a3.z; r[15] = (h16)a3.w;
    return r;
}

// Async copy of one 1KB contiguous tile (16 rows x 32 f16) into LDS.
// GVS addressing: per-lane LDS dest VGPR, per-lane 32-bit offset VGPR, SGPR64 base.
// Each issue moves 32 lanes x 16B = 512B; two issues per tile. Tracked by ASYNCcnt.
__device__ __forceinline__ void async_tile_1k(uint32_t lds_base, const h16* gbase, int lane) {
    uint32_t off = (uint32_t)lane * 16u;
    asm volatile("global_load_async_to_lds_b128 %0, %1, %2 offset:0"
                 :: "v"(lds_base + off), "v"(off), "s"(gbase) : "memory");
    asm volatile("global_load_async_to_lds_b128 %0, %1, %2 offset:512"
                 :: "v"(lds_base + off), "v"(off), "s"(gbase) : "memory");
}

// ---------------------------------------------------------------------------
// Kernel 0: transpose + f16-convert the 4 weight matrices [in,out] -> [out,in]
// so GEMM B-fragments become two b128 loads per lane.
// ---------------------------------------------------------------------------
__global__ __launch_bounds__(256) void prep_weights_kernel(
    const float* __restrict__ wq, const float* __restrict__ wk,
    const float* __restrict__ wv, const float* __restrict__ wfc,
    h16* __restrict__ wt)
{
    int m = blockIdx.y;
    const float* W = (m == 0) ? wq : (m == 1) ? wk : (m == 2) ? wv : wfc;
    int idx = blockIdx.x * 256 + threadIdx.x;     // 0..65535
    int o = idx >> 8, i = idx & 255;
    wt[(size_t)m * DD * DD + (size_t)o * DD + i] = (h16)W[(size_t)i * DD + o];
}

// ---------------------------------------------------------------------------
// Kernel 1: QKV projections. X[4096,256](f32) @ W -> f16 head-major outputs.
// q,k -> [bh][s][dk];  v -> transposed [bh][dv][s] so attn@V B-frags are b128s.
// One wave per 16x16 tile, 8 chained WMMAs over K=256.
// ---------------------------------------------------------------------------
__global__ __launch_bounds__(128) void qkv_proj_kernel(
    const float* __restrict__ q, const float* __restrict__ k, const float* __restrict__ v,
    const h16* __restrict__ wt,
    h16* __restrict__ qh, h16* __restrict__ kh, h16* __restrict__ vt)
{
    int gwave = (blockIdx.x * blockDim.x + threadIdx.x) >> 5;
    int lane  = threadIdx.x & 31;
    int mat   = gwave / 4096;          // 0=q,1=k,2=v
    int tile  = gwave % 4096;          // 256 Mtiles x 16 Ntiles
    int mt = tile >> 4, nt = tile & 15;
    const float* X = (mat == 0) ? q : (mat == 1) ? k : v;

    int idx = lane & 15, half = lane >> 4;
    int m0 = mt * 16, n0 = nt * 16;

    const float* arow = X + (size_t)(m0 + idx) * DD;
    const h16*   brow = wt + (size_t)mat * DD * DD + (size_t)(n0 + idx) * DD;

    v8f c = {};
#pragma unroll
    for (int ks = 0; ks < DD; ks += 32) {
        v16h a = frag_f32(arow + ks, half);
        v16h b = frag_h16(brow + ks, half);
        c = __builtin_amdgcn_wmma_f32_16x16x32_f16(false, a, false, b,
                                                   (short)0, c, false, false);
    }
    // scatter accumulator: elem r -> (M = r + 8*half, N = lane&15)
#pragma unroll
    for (int r = 0; r < 8; ++r) {
        int gr = m0 + r + 8 * half;          // b*S + s
        int b_ = gr >> 11, s_ = gr & (SS - 1);
        int cg = n0 + idx;
        int h_ = cg >> 5, dk = cg & 31;
        h16 val = (h16)c[r];
        if (mat == 0)
            qh[((size_t)(b_ * HH + h_) * SS + s_) * DKV + dk] = val;
        else if (mat == 1)
            kh[((size_t)(b_ * HH + h_) * SS + s_) * DKV + dk] = val;
        else
            vt[((size_t)(b_ * HH + h_) * DKV + dk) * SS + s_] = val;
    }
}

// ---------------------------------------------------------------------------
// Kernel 2: attention for one (b,h, 16-query-row tile).
//  - K tiles double-buffered into LDS via global_load_async_to_lds_b128
//  - scores stripe 16x2048 f32 in LDS, shuffle softmax
//  - bias read / attn write through the non-temporal path (512 MB stream)
//  - P@V WMMA with ds_add_f32 cross-wave reduction
// ---------------------------------------------------------------------------
__global__ __launch_bounds__(256) void attn_kernel(
    const h16* __restrict__ qh, const h16* __restrict__ kh, const h16* __restrict__ vt,
    const float* __restrict__ bias, float* __restrict__ attn_out, h16* __restrict__ oh)
{
    __shared__ float sc[16][SS];              // 128 KB
    __shared__ float ored[16][DKV];           // 2 KB
    __shared__ h16   kst[8][2][16 * DKV];     // 16 KB: per-wave double buffer

    int bh  = blockIdx.y;             // b*H + h
    int qt  = blockIdx.x;             // query tile
    int tid = threadIdx.x;
    int wave = tid >> 5, lane = tid & 31;
    int idx = lane & 15, half = lane >> 4;
    int q0 = qt * 16;

    const h16* qbase = qh + ((size_t)bh * SS + q0) * DKV;
    const h16* kbase = kh + (size_t)bh * SS * DKV;
    const h16* vbase = vt + (size_t)bh * DKV * SS;

    for (int i = tid; i < 16 * DKV; i += 256) ((float*)ored)[i] = 0.f;

    // Q fragment (A: 16x32), two b128 loads
    v16h aq = frag_h16(qbase + (size_t)idx * DKV, half);

    const float scale = 0.17677669529663687f;   // 1/sqrt(32)
    const float* brow = bias + ((size_t)bh * SS + q0) * SS;

    uint32_t kst0 = (uint32_t)(size_t)&kst[wave][0][0];
    uint32_t kst1 = (uint32_t)(size_t)&kst[wave][1][0];

    // prologue: fetch first tile for this wave (tiles kt = wave + 8*i)
    async_tile_1k(kst0, kbase + (size_t)wave * 16 * DKV, lane);

    for (int i = 0; i < 16; ++i) {
        int kt = wave + 8 * i;
        int k0 = kt * 16;
        if (i < 15) {
            uint32_t nb = (i & 1) ? kst0 : kst1;
            async_tile_1k(nb, kbase + (size_t)(kt + 8) * 16 * DKV, lane);
            asm volatile("s_wait_asynccnt 2" ::: "memory");  // current tile done
        } else {
            asm volatile("s_wait_asynccnt 0" ::: "memory");
        }
        const h16* tp = &kst[wave][i & 1][0];
        // B = Kh^T (32x16): lane holds column n = k-row (k0+idx); b128 LDS reads
        v16h bf = frag_h16(tp + (size_t)idx * DKV, half);
        v8f c = {};
        c = __builtin_amdgcn_wmma_f32_16x16x32_f16(false, aq, false, bf,
                                                   (short)0, c, false, false);
#pragma unroll
        for (int r = 0; r < 8; ++r) {
            int m = r + 8 * half, n = idx;
            float bv = __builtin_nontemporal_load(&brow[(size_t)m * SS + k0 + n]);
            sc[m][k0 + n] = c[r] * scale + bv;
        }
    }
    __syncthreads();

    // softmax: wave handles rows 2*wave, 2*wave+1
#pragma unroll
    for (int rr = 0; rr < 2; ++rr) {
        int m = wave * 2 + rr;
        float mx = -3.4e38f;
        for (int cc = lane; cc < SS; cc += 32) mx = fmaxf(mx, sc[m][cc]);
#pragma unroll
        for (int o = 16; o > 0; o >>= 1) mx = fmaxf(mx, __shfl_xor(mx, o, 32));
        float sum = 0.f;
        for (int cc = lane; cc < SS; cc += 32) {
            float e = __expf(sc[m][cc] - mx);
            sc[m][cc] = e;
            sum += e;
        }
#pragma unroll
        for (int o = 16; o > 0; o >>= 1) sum += __shfl_xor(sum, o, 32);
        float inv = 1.f / sum;
        float* arow = attn_out + ((size_t)bh * SS + q0 + m) * SS;
        for (int cc = lane; cc < SS; cc += 32) {
            float p = sc[m][cc] * inv;
            sc[m][cc] = p;
            __builtin_nontemporal_store(p, &arow[cc]);   // 256 MB stream, bypass L2 RT
        }
    }
    __syncthreads();

    // out = P[16 x 2048] @ V[2048 x 32]; wave w handles ksteps [8w, 8w+8)
    v8f acc0 = {}, acc1 = {};
    for (int ks = wave * 8; ks < wave * 8 + 8; ++ks) {
        int kb = ks * 32;
        // A = P tile (16x32) from LDS f32 -> f16 (4x ds_load_b128 per lane)
        v16h ap = frag_f32(&sc[idx][kb], half);
        // B = V tile: vt is [dv][s] so per-lane K-runs are contiguous (b128 loads)
        v16h b0 = frag_h16(vbase + (size_t)idx * SS + kb, half);
        v16h b1 = frag_h16(vbase + (size_t)(idx + 16) * SS + kb, half);
        acc0 = __builtin_amdgcn_wmma_f32_16x16x32_f16(false, ap, false, b0,
                                                      (short)0, acc0, false, false);
        acc1 = __builtin_amdgcn_wmma_f32_16x16x32_f16(false, ap, false, b1,
                                                      (short)0, acc1, false, false);
    }
#pragma unroll
    for (int r = 0; r < 8; ++r) {
        int m = r + 8 * half, n = idx;
        atomicAdd(&ored[m][n], acc0[r]);        // ds_add_f32
        atomicAdd(&ored[m][n + 16], acc1[r]);
    }
    __syncthreads();

    int b_ = bh >> 3, h_ = bh & 7;
    for (int i = tid; i < 512; i += 256) {
        int m = i >> 5, dv = i & 31;
        oh[((size_t)b_ * SS + q0 + m) * DD + h_ * DKV + dv] = (h16)ored[m][dv];
    }
}

// ---------------------------------------------------------------------------
// Kernel 3: fc GEMM + residual + LayerNorm. One block per 16-row tile.
// ---------------------------------------------------------------------------
__global__ __launch_bounds__(256) void fc_ln_kernel(
    const h16* __restrict__ oh, const h16* __restrict__ wtfc,
    const float* __restrict__ qin, const float* __restrict__ gamma,
    const float* __restrict__ beta, float* __restrict__ out)
{
    __shared__ float buf[16][DD];
    int mt = blockIdx.x;
    int tid = threadIdx.x, wave = tid >> 5, lane = tid & 31;
    int idx = lane & 15, half = lane >> 4;
    int m0 = mt * 16;

    const h16* arow = oh + (size_t)(m0 + idx) * DD;
#pragma unroll
    for (int nti = 0; nti < 2; ++nti) {
        int n0 = (wave * 2 + nti) * 16;
        const h16* brow = wtfc + (size_t)(n0 + idx) * DD;
        v8f c = {};
#pragma unroll
        for (int ks = 0; ks < DD; ks += 32) {
            v16h a = frag_h16(arow + ks, half);
            v16h b = frag_h16(brow + ks, half);
            c = __builtin_amdgcn_wmma_f32_16x16x32_f16(false, a, false, b,
                                                       (short)0, c, false, false);
        }
#pragma unroll
        for (int r = 0; r < 8; ++r)
            buf[r + 8 * half][n0 + idx] = c[r];
    }
    __syncthreads();

    // residual + LayerNorm: wave handles rows 2*wave, 2*wave+1
#pragma unroll
    for (int rr = 0; rr < 2; ++rr) {
        int m = wave * 2 + rr;
        const float* qrow = qin + (size_t)(m0 + m) * DD;
        float vals[8];
        float s1 = 0.f;
#pragma unroll
        for (int j = 0; j < 8; ++j) {
            int cc = lane + j * 32;
            vals[j] = buf[m][cc] + qrow[cc];
            s1 += vals[j];
        }
#pragma unroll
        for (int o = 16; o > 0; o >>= 1) s1 += __shfl_xor(s1, o, 32);
        float mu = s1 * (1.f / DD);
        float s2 = 0.f;
#pragma unroll
        for (int j = 0; j < 8; ++j) { float d = vals[j] - mu; s2 += d * d; }
#pragma unroll
        for (int o = 16; o > 0; o >>= 1) s2 += __shfl_xor(s2, o, 32);
        float rstd = rsqrtf(s2 * (1.f / DD) + LN_EPS);
        float* orow = out + (size_t)(m0 + m) * DD;
#pragma unroll
        for (int j = 0; j < 8; ++j) {
            int cc = lane + j * 32;
            orow[cc] = (vals[j] - mu) * rstd * gamma[cc] + beta[cc];
        }
    }
}

extern "C" void kernel_launch(void* const* d_in, const int* in_sizes, int n_in,
                              void* d_out, int out_size, void* d_ws, size_t ws_size,
                              hipStream_t stream) {
    const float* q    = (const float*)d_in[0];
    const float* k    = (const float*)d_in[1];
    const float* v    = (const float*)d_in[2];
    const float* bias = (const float*)d_in[3];
    const float* wq   = (const float*)d_in[4];
    const float* wk   = (const float*)d_in[5];
    const float* wv   = (const float*)d_in[6];
    const float* wfc  = (const float*)d_in[7];
    const float* gam  = (const float*)d_in[8];
    const float* bet  = (const float*)d_in[9];

    // workspace layout (f16): qh | kh | vt | oh (1Mi halves each) | wt (4x64Ki)
    h16* ws = (h16*)d_ws;
    const size_t HSZ = (size_t)BB * HH * SS * DKV;   // 1,048,576
    h16* qh = ws;
    h16* kh = ws + HSZ;
    h16* vt = ws + 2 * HSZ;
    h16* oh = ws + 3 * HSZ;
    h16* wt = ws + 4 * HSZ;                           // 4 * 65536 halves

    float* out  = (float*)d_out;                      // [B*S, D]
    float* attn = out + (size_t)BB * SS * DD;         // [B,H,S,S]

    prep_weights_kernel<<<dim3(DD * DD / 256, 4), 256, 0, stream>>>(wq, wk, wv, wfc, wt);
    qkv_proj_kernel<<<3 * 4096 / 4, 128, 0, stream>>>(q, k, v, wt, qh, kh, vt);
    attn_kernel<<<dim3(SS / 16, BB * HH), 256, 0, stream>>>(qh, kh, vt, bias, attn, oh);
    fc_ln_kernel<<<(BB * SS) / 16, 256, 0, stream>>>(oh, wt + 3 * (size_t)DD * DD,
                                                     q, gam, bet, out);
}
